// _DeformBlock_33878702031094
// MI455X (gfx1250) — matrix operand order repaired and checked by reading, not compiled, not run on hardware
//
#include <hip/hip_runtime.h>

// ---------------------------------------------------------------------------
// CDNA5 / gfx1250 deformable attention block.
// GEMMs: v_wmma_f32_16x16x32_f16, 16x64 strip per wave (4 accumulators,
// A-fragment reuse), B slab staged into LDS once per block via the Tensor
// Data Mover (tensor_load_to_lds + s_wait_tensorcnt). Sampling and LayerNorm
// stay on VALU. f32 -> f16 conversion once; accumulation stays f32.
// ---------------------------------------------------------------------------

typedef _Float16     v8h   __attribute__((ext_vector_type(8)));
typedef _Float16     v16h  __attribute__((ext_vector_type(16)));
typedef float        v8f   __attribute__((ext_vector_type(8)));
typedef unsigned int u32x4 __attribute__((ext_vector_type(4)));
typedef int          i32x4 __attribute__((ext_vector_type(4)));
typedef int          i32x8 __attribute__((ext_vector_type(8)));

#define NQ_    22500
#define TOTAL_ 13294
#define EMBED_ 256
#define FFN_   512

// ------------------------------ converts -----------------------------------

__global__ void k_f32_to_f16(const float* __restrict__ in,
                             _Float16* __restrict__ out, int n) {
  int i = blockIdx.x * blockDim.x + threadIdx.x;
  if (i < n) out[i] = (_Float16)in[i];
}

// W is K x N row-major (f32). Wt becomes N x K row-major (f16).
__global__ void k_transpose_f16(const float* __restrict__ W,
                                _Float16* __restrict__ Wt, int K, int N) {
  int i = blockIdx.x * blockDim.x + threadIdx.x;
  if (i < K * N) {
    int k = i / N, n = i % N;
    Wt[(size_t)n * K + k] = (_Float16)W[i];
  }
}

// --------------------- TDM stage: B slab global -> LDS ----------------------
// Copies 64 rows x K halves of Bt (row stride K) starting at row n0 into sB.
// D# layout per CDNA5 ISA (2D tensor, groups 0/1;
// groups 2/3 zero => dims 2+ unused).

template <int K>
__device__ __forceinline__ void stage_B(const _Float16* __restrict__ Bt,
                                        int n0, int Ntot,
                                        _Float16* __restrict__ sB) {
#if __has_builtin(__builtin_amdgcn_tensor_load_to_lds)
  if (threadIdx.x < 32) {  // one wave issues the DMA (EXEC ignored by TDM)
    unsigned long long ga = (unsigned long long)(Bt + (size_t)n0 * K);
    unsigned int lds = (unsigned int)(unsigned long long)(void*)sB;
    u32x4 g0 = { 1u,                                   // count=1, user mode
                 lds,                                  // lds_addr
                 (unsigned int)ga,                     // global_addr[31:0]
                 (unsigned int)((ga >> 32) & 0x1FFFFFFu) | (2u << 30) }; // +type=2
    i32x8 g1 = { (int)(1u << 16),                               // data_size = 2B
                 (int)(((unsigned)K & 0xFFFFu) << 16),          // tensor_dim0 lo
                 (int)((((unsigned)K >> 16) & 0xFFFFu) |
                       (((unsigned)Ntot & 0xFFFFu) << 16)),     // dim0 hi | dim1 lo
                 (int)((((unsigned)Ntot >> 16) & 0xFFFFu) |
                       (((unsigned)K & 0xFFFFu) << 16)),        // dim1 hi | tile_dim0=K
                 64,                                            // tile_dim1=64, tile_dim2=0
                 (int)K,                                        // tensor_dim0_stride lo
                 0, 0 };
    i32x4 z4 = {0, 0, 0, 0};
#if __clang_major__ >= 23
    i32x8 z8 = {0, 0, 0, 0, 0, 0, 0, 0};
    __builtin_amdgcn_tensor_load_to_lds(g0, g1, z4, z4, z8, 0);
#else
    __builtin_amdgcn_tensor_load_to_lds(g0, g1, z4, z4, 0);
#endif
#if __has_builtin(__builtin_amdgcn_s_wait_tensorcnt)
    __builtin_amdgcn_s_wait_tensorcnt(0);
#endif
  }
#else
  // cooperative fallback
  for (int idx = threadIdx.x; idx < 64 * (K / 8); idx += 128) {
    int r = idx / (K / 8), c = (idx % (K / 8)) * 8;
    *(v8h*)&sB[r * K + c] = *(const v8h*)&Bt[(size_t)(n0 + r) * K + c];
  }
  (void)Ntot;
#endif
  __syncthreads();
}

// ------------------------------ WMMA GEMM ----------------------------------
// C(MxN) = A(MxK f16 row-major) * Bt(NxK f16 row-major)^T + bias
// Block = 128 threads (4 waves) -> 64(M) x 64(N) tile. Wave w covers rows
// [blockIdx.x*64 + w*16, +16) and all 64 columns (4 accumulators, A reuse x4).
// B comes from the LDS slab. Layouts per CDNA5 ISA 7.12.2 (wave32):
//   A lane l: row = l&15, halves 0..7 = K kb..kb+7, 8..15 = kb+16..kb+23, kb=(l>>4)*8
//   B lane l: col = l&15, halves 0..7 = K kb..kb+7, 8..15 = kb+16..kb+23, kb=(l>>4)*8
//   D lane l: col = l&15, acc[r] = row r + 8*(l>>4)

template <int K>
__global__ void k_gemm_wmma(const _Float16* __restrict__ A,
                            const _Float16* __restrict__ Bt,
                            const float* __restrict__ bias,
                            float* __restrict__ C32,
                            _Float16* __restrict__ C16,
                            int M, int N, int relu) {
  __shared__ _Float16 sB[64 * K];

  const int lane = threadIdx.x & 31;
  const int wib  = threadIdx.x >> 5;
  const int hi   = lane >> 4;
  const int l15  = lane & 15;
  const int n0   = blockIdx.y * 64;

  stage_B<K>(Bt, n0, N, sB);

  const int mBase = blockIdx.x * 64 + wib * 16;
  int m = mBase + l15;
  if (m >= M) m = M - 1;              // clamp loads; stores are guarded

  const _Float16* Ab = A + (size_t)m * K + hi * 8;

  v8f acc[4] = {};
#pragma unroll
  for (int kk = 0; kk < K; kk += 32) {
    v8h a0 = *(const v8h*)(Ab + kk);
    v8h a1 = *(const v8h*)(Ab + kk + 16);
    __builtin_prefetch(Ab + kk + 32, 0, 0);   // global_prefetch_b8
    v16h av = __builtin_shufflevector(a0, a1, 0,1,2,3,4,5,6,7,8,9,10,11,12,13,14,15);
#pragma unroll
    for (int nt = 0; nt < 4; ++nt) {
      const _Float16* bp = &sB[(nt * 16 + l15) * K + kk + hi * 8];
      v8h b0 = *(const v8h*)bp;
      v8h b1 = *(const v8h*)(bp + 16);
      v16h bv = __builtin_shufflevector(b0, b1, 0,1,2,3,4,5,6,7,8,9,10,11,12,13,14,15);
      acc[nt] = __builtin_amdgcn_wmma_f32_16x16x32_f16(
          false, av, false, bv, (short)0, acc[nt], false, false);
    }
  }

#pragma unroll
  for (int nt = 0; nt < 4; ++nt) {
    const int n = n0 + nt * 16 + l15;
    const float bn = bias ? bias[n] : 0.f;
#pragma unroll
    for (int r = 0; r < 8; ++r) {
      int row = mBase + r + hi * 8;
      if (row < M) {
        float v = acc[nt][r] + bn;
        if (relu) v = fmaxf(v, 0.f);
        size_t o = (size_t)row * N + n;
        if (C32) C32[o] = v;
        if (C16) C16[o] = (_Float16)v;
      }
    }
  }
}

// --------------------------- bilinear sampling -----------------------------
// One wave per (query, head); lane = channel d (coalesced 128B gathers).

__device__ __forceinline__ float fetch_v(const float* __restrict__ v, int st,
                                         int H, int W, int yi, int xi,
                                         int h, int lane) {
  if (yi < 0 || yi >= H || xi < 0 || xi >= W) return 0.f;
  return v[(size_t)(st + yi * W + xi) * EMBED_ + h * 32 + lane];
}

__global__ void k_sample(const float* __restrict__ vproj,
                         const float* __restrict__ raw_off,
                         const float* __restrict__ raw_attn,
                         const float* __restrict__ ref_pts,
                         _Float16* __restrict__ attn_pre) {
  const int tid  = blockIdx.x * blockDim.x + threadIdx.x;
  const int wave = tid >> 5;
  const int lane = tid & 31;
  const int q = wave >> 3;
  const int h = wave & 7;
  if (q >= NQ_) return;

  const float* al = raw_attn + (size_t)q * 128 + h * 16;
  float w[16];
  float mx = -1e30f;
#pragma unroll
  for (int j = 0; j < 16; ++j) { w[j] = al[j]; mx = fmaxf(mx, w[j]); }
  float sum = 0.f;
#pragma unroll
  for (int j = 0; j < 16; ++j) { w[j] = __expf(w[j] - mx); sum += w[j]; }
  const float inv = 1.f / sum;

  const float* off = raw_off + (size_t)q * EMBED_ + h * 32;
  const float* ref = ref_pts + (size_t)q * 8;

  const int Hs[4] = {100, 50, 25, 13};
  const int Ws[4] = {100, 50, 25, 13};
  const int ST[4] = {0, 10000, 12500, 13125};

  float acc = 0.f;
#pragma unroll
  for (int lvl = 0; lvl < 4; ++lvl) {
    const int   Hl = Hs[lvl], Wl = Ws[lvl], st = ST[lvl];
    const float Hf = (float)Hl, Wf = (float)Wl;
    const float rx = ref[lvl * 2 + 0], ry = ref[lvl * 2 + 1];
#pragma unroll
    for (int p = 0; p < 4; ++p) {
      const float ox = off[lvl * 8 + p * 2 + 0];
      const float oy = off[lvl * 8 + p * 2 + 1];
      const float x = (rx + ox / Wf) * Wf - 0.5f;
      const float y = (ry + oy / Hf) * Hf - 0.5f;
      const float x0f = floorf(x), y0f = floorf(y);
      const float dx = x - x0f, dy = y - y0f;
      const int x0 = (int)x0f, y0 = (int)y0f;
      const float g00 = fetch_v(vproj, st, Hl, Wl, y0,     x0,     h, lane);
      const float g10 = fetch_v(vproj, st, Hl, Wl, y0,     x0 + 1, h, lane);
      const float g01 = fetch_v(vproj, st, Hl, Wl, y0 + 1, x0,     h, lane);
      const float g11 = fetch_v(vproj, st, Hl, Wl, y0 + 1, x0 + 1, h, lane);
      const float bil = g00 * (1.f - dx) * (1.f - dy) + g10 * dx * (1.f - dy)
                      + g01 * (1.f - dx) * dy         + g11 * dx * dy;
      acc += w[lvl * 4 + p] * inv * bil;
    }
  }
  attn_pre[(size_t)q * EMBED_ + h * 32 + lane] = (_Float16)acc;
}

// --------------------------- fused add + LayerNorm --------------------------

__global__ void k_add_ln(const float* __restrict__ a, const float* __restrict__ b,
                         const float* __restrict__ g, const float* __restrict__ bet,
                         float* __restrict__ out32, _Float16* __restrict__ out16) {
  __shared__ float sh[EMBED_];
  const int row = blockIdx.x, tid = threadIdx.x;
  const size_t base = (size_t)row * EMBED_ + tid;
  const float v = a[base] + b[base];
  sh[tid] = v;
  __syncthreads();
  for (int s = 128; s > 0; s >>= 1) { if (tid < s) sh[tid] += sh[tid + s]; __syncthreads(); }
  const float mean = sh[0] * (1.f / EMBED_);
  __syncthreads();
  const float d = v - mean;
  sh[tid] = d * d;
  __syncthreads();
  for (int s = 128; s > 0; s >>= 1) { if (tid < s) sh[tid] += sh[tid + s]; __syncthreads(); }
  const float var = sh[0] * (1.f / EMBED_);
  const float r = rsqrtf(var + 1e-5f);
  const float o = d * r * g[tid] + bet[tid];
  if (out32) out32[base] = o;
  if (out16) out16[base] = (_Float16)o;
}

// ------------------------------- launcher ----------------------------------

extern "C" void kernel_launch(void* const* d_in, const int* in_sizes, int n_in,
                              void* d_out, int out_size, void* d_ws, size_t ws_size,
                              hipStream_t stream) {
  (void)in_sizes; (void)n_in; (void)out_size; (void)ws_size;

  const float* q       = (const float*)d_in[0];
  const float* ref_pts = (const float*)d_in[1];
  const float* value   = (const float*)d_in[2];
  const float* Wv    = (const float*)d_in[5];
  const float* bv    = (const float*)d_in[6];
  const float* Woff  = (const float*)d_in[7];
  const float* boff  = (const float*)d_in[8];
  const float* Wattn = (const float*)d_in[9];
  const float* battn = (const float*)d_in[10];
  const float* Wout  = (const float*)d_in[11];
  const float* bout  = (const float*)d_in[12];
  const float* g1    = (const float*)d_in[13];
  const float* b1    = (const float*)d_in[14];
  const float* W1    = (const float*)d_in[15];
  const float* bb1   = (const float*)d_in[16];
  const float* W2    = (const float*)d_in[17];
  const float* bb2   = (const float*)d_in[18];
  const float* g2    = (const float*)d_in[19];
  const float* b2    = (const float*)d_in[20];
  float* out = (float*)d_out;

  // ---- workspace layout (reuse chains are ordered on the single stream) ----
  char* ws = (char*)d_ws;
  size_t off = 0;
  auto alloc = [&](size_t bytes) -> void* {
    void* p = ws + off;
    off += (bytes + 255) & ~(size_t)255;
    return p;
  };
  _Float16* q16      = (_Float16*)alloc((size_t)NQ_ * EMBED_ * 2);   // -> attn_pre -> x16
  _Float16* attn_pre = q16;
  _Float16* x16      = q16;
  _Float16* val16    = (_Float16*)alloc((size_t)TOTAL_ * EMBED_ * 2);
  float*    vproj    = (float*)alloc((size_t)TOTAL_ * EMBED_ * 4);
  float*    r3       = (float*)alloc((size_t)NQ_ * EMBED_ * 4);      // raw_off -> attn_out -> y
  float*    raw_attn = (float*)alloc((size_t)NQ_ * 128 * 4);
  float*    x        = (float*)alloc((size_t)NQ_ * EMBED_ * 4);
  _Float16* h16      = (_Float16*)alloc((size_t)NQ_ * FFN_ * 2);
  _Float16* WvT      = (_Float16*)alloc(256 * 256 * 2);
  _Float16* WoffT    = (_Float16*)alloc(256 * 256 * 2);
  _Float16* WattnT   = (_Float16*)alloc(128 * 256 * 2);
  _Float16* WoutT    = (_Float16*)alloc(256 * 256 * 2);
  _Float16* W1T      = (_Float16*)alloc(512 * 256 * 2);
  _Float16* W2T      = (_Float16*)alloc(256 * 512 * 2);

  const int mtQ = (NQ_ + 63) / 64;     // 352
  const int mtV = (TOTAL_ + 63) / 64;  // 208

  // ---- stage 0: precision conversion + weight transposes ----
  {
    int n = NQ_ * EMBED_;
    k_f32_to_f16<<<(n + 255) / 256, 256, 0, stream>>>(q, q16, n);
    n = TOTAL_ * EMBED_;
    k_f32_to_f16<<<(n + 255) / 256, 256, 0, stream>>>(value, val16, n);
  }
  k_transpose_f16<<<(256 * 256 + 255) / 256, 256, 0, stream>>>(Wv,    WvT,    256, 256);
  k_transpose_f16<<<(256 * 256 + 255) / 256, 256, 0, stream>>>(Woff,  WoffT,  256, 256);
  k_transpose_f16<<<(256 * 128 + 255) / 256, 256, 0, stream>>>(Wattn, WattnT, 256, 128);
  k_transpose_f16<<<(256 * 256 + 255) / 256, 256, 0, stream>>>(Wout,  WoutT,  256, 256);
  k_transpose_f16<<<(256 * 512 + 255) / 256, 256, 0, stream>>>(W1,    W1T,    256, 512);
  k_transpose_f16<<<(512 * 256 + 255) / 256, 256, 0, stream>>>(W2,    W2T,    512, 256);

  // ---- stage 1: value projection v = value @ Wv + bv (f32 out for sampling)
  k_gemm_wmma<256><<<dim3(mtV, 4), 128, 0, stream>>>(val16, WvT, bv, vproj, nullptr,
                                                     TOTAL_, 256, 0);
  // ---- stage 2: offsets + attention logits
  k_gemm_wmma<256><<<dim3(mtQ, 4), 128, 0, stream>>>(q16, WoffT, boff, r3, nullptr,
                                                     NQ_, 256, 0);
  k_gemm_wmma<256><<<dim3(mtQ, 2), 128, 0, stream>>>(q16, WattnT, battn, raw_attn, nullptr,
                                                     NQ_, 128, 0);
  // ---- stage 3: softmax + bilinear deformable sampling (f16 out = GEMM A)
  k_sample<<<NQ_, 256, 0, stream>>>(vproj, r3, raw_attn, ref_pts, attn_pre);
  // ---- stage 4: output projection
  k_gemm_wmma<256><<<dim3(mtQ, 4), 128, 0, stream>>>(attn_pre, WoutT, bout, r3, nullptr,
                                                     NQ_, 256, 0);
  // ---- stage 5: x = LN(q + attn) ; also x in f16 for the FFN
  k_add_ln<<<NQ_, 256, 0, stream>>>(q, r3, g1, b1, x, x16);
  // ---- stage 6: h = relu(x @ W1 + bb1) (f16 out)
  k_gemm_wmma<256><<<dim3(mtQ, 8), 128, 0, stream>>>(x16, W1T, bb1, nullptr, h16,
                                                     NQ_, 512, 1);
  // ---- stage 7: y = h @ W2 + bb2
  k_gemm_wmma<512><<<dim3(mtQ, 4), 128, 0, stream>>>(h16, W2T, bb2, r3, nullptr,
                                                     NQ_, 256, 0);
  // ---- stage 8: out = LN(x + y)
  k_add_ln<<<NQ_, 256, 0, stream>>>(x, r3, g2, b2, out, nullptr);
}
